// GCNLayer_15375982920434
// MI455X (gfx1250) — compile-verified, matrix operand order checked by reading, not run on hardware
//
#include <hip/hip_runtime.h>

typedef __attribute__((ext_vector_type(16))) __bf16 v16bf;
typedef __attribute__((ext_vector_type(8)))  __bf16 v8bf;
typedef __attribute__((ext_vector_type(8)))  float  v8f;
typedef __attribute__((ext_vector_type(4)))  float  v4f;

#define IN_FEATS  256
#define OUT_FEATS 64

// ---------------------------------------------------------------------------
// Kernel 1: XW = X @ W via bf16 WMMA, fp32 accumulate.
// Block = 256 threads = 8 waves; each wave computes a 16x64 output strip.
// W (256x64 fp32) is converted + transposed into LDS as WT[n][k] bf16 (32 KB)
// so B-fragment loads are contiguous ds_load_b128.
// ---------------------------------------------------------------------------
__global__ __launch_bounds__(256) void gcn_gemm_wmma(
    const float* __restrict__ X, const float* __restrict__ W,
    float* __restrict__ XW, int N)
{
    __shared__ __bf16 WT[OUT_FEATS * IN_FEATS];   // [n][k], 32 KB

    const int tid = threadIdx.x;

    // Cooperative W load: fp32 -> bf16, transpose [k][n] -> [n][k]
    for (int i = tid; i < IN_FEATS * OUT_FEATS; i += 256) {
        int k = i >> 6;     // / OUT_FEATS
        int n = i & 63;     // % OUT_FEATS
        WT[n * IN_FEATS + k] = (__bf16)W[i];
    }
    __syncthreads();

    const int wave    = tid >> 5;           // 0..7
    const int lane    = tid & 31;
    const int halfsel = lane >> 4;          // 0 or 1
    const int lrow    = lane & 15;          // A-matrix row within tile
    const int col0    = lane & 15;          // B/D column within tile
    const int khalf   = halfsel * 8;

    const int rowBase = blockIdx.x * 128 + wave * 16;
    const int arow    = rowBase + lrow;
    const int arowLd  = arow < N ? arow : (N - 1);   // clamp, keep EXEC full
    const float* xr   = X + (size_t)arowLd * IN_FEATS;

    v8f acc[4];
    #pragma unroll
    for (int t = 0; t < 4; ++t) acc[t] = (v8f){};

    #pragma unroll
    for (int kc = 0; kc < 8; ++kc) {
        const int kb = kc * 32 + khalf;

        // A fragment: this lane needs K = kb..kb+7 and kb+16..kb+23 (fp32->bf16)
        v4f x0 = *(const v4f*)(xr + kb);
        v4f x1 = *(const v4f*)(xr + kb + 4);
        v4f x2 = *(const v4f*)(xr + kb + 16);
        v4f x3 = *(const v4f*)(xr + kb + 20);
        v16bf a;
        #pragma unroll
        for (int j = 0; j < 4; ++j) {
            a[j]      = (__bf16)x0[j];
            a[4 + j]  = (__bf16)x1[j];
            a[8 + j]  = (__bf16)x2[j];
            a[12 + j] = (__bf16)x3[j];
        }

        // 4 column tiles of B from LDS (contiguous in WT -> ds_load_b128)
        #pragma unroll
        for (int t = 0; t < 4; ++t) {
            const __bf16* wp = &WT[(t * 16 + col0) * IN_FEATS + kb];
            v8bf b0 = *(const v8bf*)(wp);
            v8bf b1 = *(const v8bf*)(wp + 16);
            v16bf b;
            #pragma unroll
            for (int j = 0; j < 8; ++j) { b[j] = b0[j]; b[8 + j] = b1[j]; }

            acc[t] = __builtin_amdgcn_wmma_f32_16x16x32_bf16(
                /*neg_a=*/false, a, /*neg_b=*/false, b,
                /*c_mod=*/(short)0, acc[t],
                /*reuse_a=*/false, /*reuse_b=*/false);
        }
    }

    // D layout: VGPR d -> row rowBase + d + 8*halfsel, col = t*16 + col0
    #pragma unroll
    for (int d = 0; d < 8; ++d) {
        const int srow = rowBase + d + halfsel * 8;
        if (srow < N) {
            float* orow = XW + (size_t)srow * OUT_FEATS + col0;
            #pragma unroll
            for (int t = 0; t < 4; ++t) orow[t * 16] = acc[t][d];
        }
    }
}

// ---------------------------------------------------------------------------
// Kernel 0: zero the output accumulator (float4 stores).
// ---------------------------------------------------------------------------
__global__ __launch_bounds__(256) void gcn_zero(float* __restrict__ out, int n4)
{
    int i = blockIdx.x * 256 + threadIdx.x;
    if (i < n4) ((v4f*)out)[i] = (v4f){};
}

// ---------------------------------------------------------------------------
// Kernel 2: edge scatter. 16 threads per edge, float4 gather of XW[col]
// (L2-resident, 25.6 MB), global_atomic_add_f32 scatter into out[row].
// ---------------------------------------------------------------------------
__global__ __launch_bounds__(256) void gcn_spmm_scatter(
    const int* __restrict__ erow, const int* __restrict__ ecol,
    const float* __restrict__ evals, const float* __restrict__ XW,
    float* __restrict__ out, int E)
{
    long long tid = (long long)blockIdx.x * 256 + threadIdx.x;
    int e = (int)(tid >> 4);
    if (e >= E) return;
    int f = (int)(tid & 15) << 2;

    float v = evals[e];
    int   c = ecol[e];
    int   r = erow[e];

    v4f x = *(const v4f*)(XW + (size_t)c * OUT_FEATS + f);
    float* o = out + (size_t)r * OUT_FEATS + f;
    atomicAdd(o + 0, v * x[0]);
    atomicAdd(o + 1, v * x[1]);
    atomicAdd(o + 2, v * x[2]);
    atomicAdd(o + 3, v * x[3]);
}

// ---------------------------------------------------------------------------
// Kernel 3: bias + ReLU, in place.
// ---------------------------------------------------------------------------
__global__ __launch_bounds__(256) void gcn_bias_relu(
    float* __restrict__ out, const float* __restrict__ b, int total)
{
    int i = blockIdx.x * 256 + threadIdx.x;
    if (i < total) {
        float v = out[i] + b[i & (OUT_FEATS - 1)];
        out[i] = v > 0.0f ? v : 0.0f;
    }
}

// ---------------------------------------------------------------------------
extern "C" void kernel_launch(void* const* d_in, const int* in_sizes, int n_in,
                              void* d_out, int out_size, void* d_ws, size_t ws_size,
                              hipStream_t stream)
{
    const float* X     = (const float*)d_in[0];
    const int*   erow  = (const int*)  d_in[1];
    const int*   ecol  = (const int*)  d_in[2];
    const float* evals = (const float*)d_in[3];
    const float* W     = (const float*)d_in[4];
    const float* b     = (const float*)d_in[5];
    float*       out   = (float*)d_out;
    float*       XW    = (float*)d_ws;          // N*64 fp32 = 25.6 MB scratch

    const int N = in_sizes[0] / IN_FEATS;
    const int E = in_sizes[1];
    const int total = N * OUT_FEATS;

    // 1) out = 0
    {
        int n4 = total / 4;
        gcn_zero<<<(n4 + 255) / 256, 256, 0, stream>>>(out, n4);
    }
    // 2) XW = X @ W  (bf16 WMMA)
    {
        int nb = (N + 127) / 128;
        gcn_gemm_wmma<<<nb, 256, 0, stream>>>(X, W, XW, N);
    }
    // 3) out[row] += val * XW[col]  (atomic scatter)
    {
        long long threads = (long long)E * 16;
        int nb = (int)((threads + 255) / 256);
        gcn_spmm_scatter<<<nb, 256, 0, stream>>>(erow, ecol, evals, XW, out, E);
    }
    // 4) out = relu(out + b)
    {
        gcn_bias_relu<<<(total + 255) / 256, 256, 0, stream>>>(out, b, total);
    }
}
